// QuantLlamaAttention_66803921322311
// MI455X (gfx1250) — matrix operand order repaired and checked by reading, not compiled, not run on hardware
//
#include <hip/hip_runtime.h>
#include <hip/hip_bf16.h>

typedef __attribute__((ext_vector_type(16))) _Float16 v16h;
typedef __attribute__((ext_vector_type(8)))  float    v8f;

union V16H { v16h v; uint4 u[2]; _Float16 h[16]; };

#define SLEN 2048
#define HDIM 4096
#define NHEAD 32
#define HD 128
#define QK_SCALE 0.08838834764831845f  // 1/sqrt(128)

// ---------------------------------------------------------------------------
// Kernel 0: f32 -> f16 convert (hidden_states)
// ---------------------------------------------------------------------------
__global__ __launch_bounds__(256) void cvt_f32_f16(const float* __restrict__ x,
                                                   _Float16* __restrict__ y, int n) {
  int i = blockIdx.x * 256 + threadIdx.x;
  if (i < n) y[i] = (_Float16)x[i];
}

// ---------------------------------------------------------------------------
// Fused int4-dequant GEMM, M-blocked: each wave computes MT 16-row tiles of one
// 16-column strip. B is dequantized ONCE per K-step and fed to MT WMMAs, so the
// int4->f16 VALU cost and weight re-reads are amortized MT x.
// GPTQ: W[k,n] = scales[k/G,n] * (q[k,n] - (zeros[k/G,n] + 1))
// ---------------------------------------------------------------------------
template <bool OUT_F32, int MT>
__global__ __launch_bounds__(256) void gemm_dq(const _Float16* __restrict__ A,
                                               const int* __restrict__ qw,
                                               const int* __restrict__ qz,
                                               const float* __restrict__ sc,
                                               void* __restrict__ Cout,
                                               int K, int N) {
  const int lane = threadIdx.x & 31;
  const int wave = (blockIdx.x * 256 + threadIdx.x) >> 5;
  const int ntiles = N >> 4;
  const int mblk = wave / ntiles;               // block of MT*16 rows
  const int tileN = wave - mblk * ntiles;

  const int lrow = lane & 15;
  const int klo = (lane < 16) ? 0 : 16;         // A-fragment K sub-base per half
  const int n = (tileN << 4) + lrow;            // B column this lane owns
  const int row0 = mblk * (MT * 16);

  v8f acc[MT];
#pragma unroll
  for (int mt = 0; mt < MT; ++mt) acc[mt] = (v8f){};

  for (int k0 = 0; k0 < K; k0 += 32) {
    // ---- B fragment: dequant 16 consecutive K for column n (once per step) ----
    const int kb = k0 + ((lane >> 4) << 4);
    const int p0 = qw[(size_t)(kb >> 3) * N + n];
    const int p1 = qw[(size_t)((kb >> 3) + 1) * N + n];
    const int gi = kb >> 7;                     // / G (=128)
    const int z = ((qz[(size_t)gi * (N >> 3) + (n >> 3)] >> ((n & 7) * 4)) & 0xF) + 1;
    const float s = sc[(size_t)gi * N + n];
    const float zs = (float)z * s;

    V16H bv;
#pragma unroll
    for (int e = 0; e < 8; ++e) {
      bv.h[e]     = (_Float16)((float)((p0 >> (4 * e)) & 0xF) * s - zs);
      bv.h[e + 8] = (_Float16)((float)((p1 >> (4 * e)) & 0xF) * s - zs);
    }

    // ---- MT WMMAs sharing the dequantized B ----
#pragma unroll
    for (int mt = 0; mt < MT; ++mt) {
      const _Float16* ap = A + (size_t)(row0 + mt * 16 + lrow) * K + k0 + klo;
      V16H av;
      av.u[0] = *(const uint4*)(ap);
      av.u[1] = *(const uint4*)(ap + 8);
      acc[mt] = __builtin_amdgcn_wmma_f32_16x16x32_f16(false, av.v, false, bv.v,
                                                       (short)0, acc[mt], false,
                                                       false);
    }
  }

  // ---- store C (f32 C layout: vgpr r -> row r + 8*(lane>=16), col lane&15) ----
  const int half8 = (lane >> 4) << 3;
#pragma unroll
  for (int mt = 0; mt < MT; ++mt) {
#pragma unroll
    for (int r = 0; r < 8; ++r) {
      size_t idx = (size_t)(row0 + mt * 16 + half8 + r) * N + n;
      if (OUT_F32) ((float*)Cout)[idx] = acc[mt][r];
      else         ((_Float16*)Cout)[idx] = (_Float16)acc[mt][r];
    }
  }
}

// ---------------------------------------------------------------------------
// RoPE + head layout: qkv f16 [S][3H] -> qh/kh [NH][S][HD] f16, vt [NH][HD][S]
// ---------------------------------------------------------------------------
__global__ __launch_bounds__(256) void rope_layout(const _Float16* __restrict__ qkv,
                                                   _Float16* __restrict__ qh,
                                                   _Float16* __restrict__ kh,
                                                   _Float16* __restrict__ vt) {
  int idx = blockIdx.x * 256 + threadIdx.x;     // over S*H
  int s = idx >> 12;                            // / 4096
  int hd = idx & 4095;
  int h = hd >> 7;
  int d = hd & 127;
  size_t rbase = (size_t)s * (3 * HDIM);

  float qv = (float)qkv[rbase + hd];
  float kv = (float)qkv[rbase + HDIM + hd];
  float vv = (float)qkv[rbase + 2 * HDIM + hd];
  int dp = (h << 7) + (d ^ 64);                 // rotate_half partner
  float qp = (float)qkv[rbase + dp];
  float kp = (float)qkv[rbase + HDIM + dp];

  int i = d & 63;
  float inv = __expf(-(float)i * (9.210340371976184f / 64.0f));  // 10000^(-i/64)
  float ang = (float)s * inv;
  float sn, cs;
  __sincosf(ang, &sn, &cs);
  float sign = (d < 64) ? -1.0f : 1.0f;

  qh[((size_t)h * SLEN + s) * HD + d] = (_Float16)(qv * cs + sign * qp * sn);
  kh[((size_t)h * SLEN + s) * HD + d] = (_Float16)(kv * cs + sign * kp * sn);
  vt[((size_t)h * HD + d) * SLEN + s] = (_Float16)vv;
}

// ---------------------------------------------------------------------------
// Flash attention: one wave per (head, 16-query tile). Causal, streaming
// softmax, 64 keys per iteration (4 S-tiles) to amortize softmax VALU/shfl.
// ---------------------------------------------------------------------------
__global__ __launch_bounds__(256) void attn_kernel(const _Float16* __restrict__ qh,
                                                   const _Float16* __restrict__ kh,
                                                   const _Float16* __restrict__ vt,
                                                   _Float16* __restrict__ ao) {
  __shared__ __align__(16) _Float16 plds[8][16 * 64];  // P staging, 2KB per wave

  const int wid = threadIdx.x >> 5;
  const int lane = threadIdx.x & 31;
  const int task = blockIdx.x * 8 + wid;        // NHEAD * (SLEN/16) = 4096 tasks
  const int head = task >> 7;
  const int qi = task & 127;
  const int q0 = qi << 4;

  const int lrow = lane & 15;
  const int klo = (lane < 16) ? 0 : 16;
  const int half8 = (lane >> 4) << 3;
  const int kshift = (lane >> 4) << 4;          // B-fragment K sub-base (0 or 16)

  // Q fragments over HD=128 (4 x K=32), pre-scaled by 1/sqrt(HD)
  v16h qa[4];
  {
    const _Float16* qrow = qh + ((size_t)head * SLEN + q0 + lrow) * HD;
#pragma unroll
    for (int kc = 0; kc < 4; ++kc) {
      V16H t;
      t.u[0] = *(const uint4*)(qrow + kc * 32 + klo);
      t.u[1] = *(const uint4*)(qrow + kc * 32 + klo + 8);
#pragma unroll
      for (int e = 0; e < 16; ++e) t.h[e] = t.h[e] * (_Float16)QK_SCALE;
      qa[kc] = t.v;
    }
  }

  v8f o[8];
#pragma unroll
  for (int dt = 0; dt < 8; ++dt) o[dt] = (v8f){};
  float mrow[8], lsum[8];
#pragma unroll
  for (int r = 0; r < 8; ++r) { mrow[r] = -1e30f; lsum[r] = 0.0f; }

  const int jqmax = qi >> 2;                    // 64-key groups covering keys <= q
  for (int jq = 0; jq <= jqmax; ++jq) {
    const int kb = jq << 6;

    // ---- S = Q K^T for four 16-key tiles ----
    v8f st[4];
#pragma unroll
    for (int t = 0; t < 4; ++t) st[t] = (v8f){};
#pragma unroll
    for (int kc = 0; kc < 4; ++kc) {
#pragma unroll
      for (int t = 0; t < 4; ++t) {
        const _Float16* krow =
            kh + ((size_t)head * SLEN + kb + t * 16 + lrow) * HD + kc * 32 + kshift;
        V16H b;
        b.u[0] = *(const uint4*)(krow);
        b.u[1] = *(const uint4*)(krow + 8);
        st[t] = __builtin_amdgcn_wmma_f32_16x16x32_f16(false, qa[kc], false, b.v,
                                                       (short)0, st[t], false,
                                                       false);
      }
    }

    // ---- causal mask + streaming softmax (rows live across 16-lane halves) ----
#pragma unroll
    for (int r = 0; r < 8; ++r) {
      const int qg = q0 + half8 + r;
      float v0 = (kb + lrow <= qg)      ? st[0][r] : -1e30f;
      float v1 = (kb + 16 + lrow <= qg) ? st[1][r] : -1e30f;
      float v2 = (kb + 32 + lrow <= qg) ? st[2][r] : -1e30f;
      float v3 = (kb + 48 + lrow <= qg) ? st[3][r] : -1e30f;
      float t = fmaxf(fmaxf(v0, v1), fmaxf(v2, v3));
      t = fmaxf(t, __shfl_xor(t, 1, 32));
      t = fmaxf(t, __shfl_xor(t, 2, 32));
      t = fmaxf(t, __shfl_xor(t, 4, 32));
      t = fmaxf(t, __shfl_xor(t, 8, 32));
      float mn = fmaxf(mrow[r], t);
      float rescale = __expf(mrow[r] - mn);
      float p0 = __expf(v0 - mn);
      float p1 = __expf(v1 - mn);
      float p2 = __expf(v2 - mn);
      float p3 = __expf(v3 - mn);
      float rs = (p0 + p1) + (p2 + p3);
      rs += __shfl_xor(rs, 1, 32);
      rs += __shfl_xor(rs, 2, 32);
      rs += __shfl_xor(rs, 4, 32);
      rs += __shfl_xor(rs, 8, 32);
      lsum[r] = lsum[r] * rescale + rs;
      mrow[r] = mn;
#pragma unroll
      for (int dt = 0; dt < 8; ++dt) o[dt][r] *= rescale;
      _Float16* pr = &plds[wid][(half8 + r) * 64 + lrow];
      pr[0]  = (_Float16)p0;
      pr[16] = (_Float16)p1;
      pr[32] = (_Float16)p2;
      pr[48] = (_Float16)p3;
    }

    // per-wave LDS round-trip: DS ops are in-order per wave; wait for stores
    asm volatile("s_wait_dscnt 0" ::: "memory");

    // ---- reload P as two A fragments (keys 0-31, 32-63), O += P x V ----
    V16H pa0, pa1;
    pa0.u[0] = *(const uint4*)&plds[wid][lrow * 64 + klo];
    pa0.u[1] = *(const uint4*)&plds[wid][lrow * 64 + klo + 8];
    pa1.u[0] = *(const uint4*)&plds[wid][lrow * 64 + 32 + klo];
    pa1.u[1] = *(const uint4*)&plds[wid][lrow * 64 + 32 + klo + 8];

    const _Float16* vbase =
        vt + ((size_t)head * HD + lrow) * SLEN + kb + kshift;
#pragma unroll
    for (int dt = 0; dt < 8; ++dt) {
      const _Float16* vp = vbase + (size_t)dt * 16 * SLEN;
      V16H bv0, bv1;
      bv0.u[0] = *(const uint4*)vp;
      bv0.u[1] = *(const uint4*)(vp + 8);
      bv1.u[0] = *(const uint4*)(vp + 32);
      bv1.u[1] = *(const uint4*)(vp + 40);
      o[dt] = __builtin_amdgcn_wmma_f32_16x16x32_f16(false, pa0.v, false, bv0.v,
                                                     (short)0, o[dt], false, false);
      o[dt] = __builtin_amdgcn_wmma_f32_16x16x32_f16(false, pa1.v, false, bv1.v,
                                                     (short)0, o[dt], false, false);
    }
  }

  // ---- normalize + write attention output as f16 [S][H] for O-proj GEMM ----
#pragma unroll
  for (int dt = 0; dt < 8; ++dt) {
#pragma unroll
    for (int r = 0; r < 8; ++r) {
      float val = o[dt][r] / lsum[r];
      ao[(size_t)(q0 + half8 + r) * HDIM + head * HD + dt * 16 + lrow] =
          (_Float16)val;
    }
  }
}

// ---------------------------------------------------------------------------
// Host launch
// ---------------------------------------------------------------------------
extern "C" void kernel_launch(void* const* d_in, const int* in_sizes, int n_in,
                              void* d_out, int out_size, void* d_ws, size_t ws_size,
                              hipStream_t stream) {
  (void)in_sizes; (void)n_in; (void)out_size; (void)ws_size;
  const float* hidden = (const float*)d_in[0];
  // d_in[1] attention_mask, d_in[2] position_ids: implied by causal + arange
  const int*   qw_qkv = (const int*)d_in[3];
  const int*   qz_qkv = (const int*)d_in[4];
  const float* sc_qkv = (const float*)d_in[5];
  const int*   qw_o   = (const int*)d_in[6];
  const int*   qz_o   = (const int*)d_in[7];
  const float* sc_o   = (const float*)d_in[8];
  float* out = (float*)d_out;

  char* ws = (char*)d_ws;
  const size_t MB = 1024 * 1024;
  _Float16* x16 = (_Float16*)(ws);                    // 16 MB  [S][H]
  _Float16* qkv = (_Float16*)(ws + 16 * MB);          // 48 MB  [S][3H]
  _Float16* qh  = (_Float16*)(ws + 64 * MB);          // 16 MB  [NH][S][HD]
  _Float16* kh  = (_Float16*)(ws + 80 * MB);          // 16 MB  [NH][S][HD]
  _Float16* vt  = (_Float16*)(ws + 96 * MB);          // 16 MB  [NH][HD][S]
  _Float16* ao  = (_Float16*)(ws + 112 * MB);         // 16 MB  [S][H]

  // 1) hidden f32 -> f16
  cvt_f32_f16<<<(SLEN * HDIM) / 256, 256, 0, stream>>>(hidden, x16, SLEN * HDIM);

  // 2) QKV fused dequant GEMM: (2048x4096) x (4096x12288)
  //    wave task = 128 rows x 16 cols: (2048/128)*(12288/16) = 12288 waves
  gemm_dq<false, 8><<<(SLEN / 128) * (3 * HDIM / 16) / 8, 256, 0, stream>>>(
      x16, qw_qkv, qz_qkv, sc_qkv, (void*)qkv, HDIM, 3 * HDIM);

  // 3) RoPE + head-major layouts (V transposed for contiguous B-fragments)
  rope_layout<<<(SLEN * HDIM) / 256, 256, 0, stream>>>(qkv, qh, kh, vt);

  // 4) causal flash attention: 32 heads x 128 q-tiles = 4096 waves
  attn_kernel<<<NHEAD * (SLEN / 16) / 8, 256, 0, stream>>>(qh, kh, vt, ao);

  // 5) O projection fused dequant GEMM -> f32 output
  gemm_dq<true, 8><<<(SLEN / 128) * (HDIM / 16) / 8, 256, 0, stream>>>(
      ao, qw_o, qz_o, sc_o, (void*)out, HDIM, HDIM);
}
